// EncoderLayer_82592221102373
// MI455X (gfx1250) — compile-verified, matrix operand order checked by reading, not run on hardware
//
#include <hip/hip_runtime.h>

typedef __bf16 bf16;
typedef __attribute__((ext_vector_type(16))) __bf16 v16bf;
typedef __attribute__((ext_vector_type(8)))  float  v8f;
typedef __attribute__((ext_vector_type(4)))  unsigned int u32x4;
typedef __attribute__((ext_vector_type(8)))  int i32x8;
typedef __attribute__((ext_vector_type(4)))  int i32x4;

#define LSTR 40   // LDS row stride in bf16 elems: 32 data + 8 pad (80B, bank-conflict-free, 16B aligned)

#if __has_builtin(__builtin_amdgcn_tensor_load_to_lds)
#define HAVE_TDM 1
#else
#define HAVE_TDM 0
#endif

constexpr int Bc = 8, Sc = 1024, Dc = 768, Hc = 12, Fc = 3072;

// ---------------------------------------------------------------------------
// Fragment load from LDS. Tile stored row-major [rows][32] with LSTR stride.
// Per ISA 16-bit A layout (16x32): lane<16 holds K {0..7,16..23} of row lane,
// lane>=16 holds K {8..15,24..31} of row lane-16  -> two 16B contiguous loads.
// B operands are staged as [N][K] so the same loader works for both.
// ---------------------------------------------------------------------------
__device__ inline v16bf ld_frag(const bf16* lds, int row0, int lane) {
  const bf16* p = lds + (row0 + (lane & 15)) * LSTR + ((lane >> 4) << 3);
  union { uint4 q[2]; v16bf v; } u;
  u.q[0] = *(const uint4*)(p);
  u.q[1] = *(const uint4*)(p + 16);
  return u.v;
}

template<int ROWS>
__device__ inline void stage_bf16(bf16* lds, const bf16* g, int ld, int tid) {
  for (int c = tid; c < ROWS * 4; c += 256) {
    int r = c >> 2, c8 = (c & 3) << 3;
    *(uint4*)(lds + r * LSTR + c8) = *(const uint4*)(g + (size_t)r * ld + c8);
  }
}

template<int ROWS>
__device__ inline void stage_f32(bf16* lds, const float* g, int ld, int tid) {
  for (int c = tid; c < ROWS * 4; c += 256) {
    int r = c >> 2, c8 = (c & 3) << 3;
    const float* p = g + (size_t)r * ld + c8;
    union { bf16 h[8]; uint4 q; } t;
#pragma unroll
    for (int i = 0; i < 8; ++i) t.h[i] = (bf16)p[i];
    *(uint4*)(lds + r * LSTR + c8) = t.q;
  }
}

// ---------------------------------------------------------------------------
// TDM: DMA one [rows x 32] bf16 tile (row stride = ld elems) from global into
// LDS at byte offset lds_byte, expanding each 64B row to 80B via D# padding
// (pad_interval=3 -> every 16 DWORDs, pad_amount=3 -> 4 DWORDs = 16B).
// ---------------------------------------------------------------------------
__device__ inline unsigned lds_off(const void* p) {
  // flat shared address: low 32 bits are the LDS byte offset
  return (unsigned)(unsigned long long)(uintptr_t)p;
}

#if HAVE_TDM
__device__ inline void tdm_load_tile(unsigned lds_byte, const bf16* g, int rows, int ld) {
  unsigned long long ga = (unsigned long long)(uintptr_t)g;
  u32x4 g0;
  g0[0] = 1u;                                            // count=1, user mode
  g0[1] = lds_byte;                                      // lds_addr (bytes)
  g0[2] = (unsigned)(ga & 0xffffffffu);                  // global_addr[31:0]
  g0[3] = (unsigned)((ga >> 32) & 0x1ffffffu) | (2u << 30); // ga[56:32] | type=2
  i32x8 g1;
  g1[0] = (1 << 16) | (1 << 20) | (3 << 22) | (3 << 25); // data_size=2B, pad_en, interval, amount
  g1[1] = (32 & 0xffff) << 16;                           // tensor_dim0 = 32 (low16 @ bit48)
  g1[2] = (rows & 0xffff) << 16;                         // dim0 hi16=0 | tensor_dim1 low16
  g1[3] = (32 & 0xffff) << 16;                           // dim1 hi16=0 | tile_dim0 = 32
  g1[4] = (rows & 0xffff);                               // tile_dim1 = rows, tile_dim2 = 0
  g1[5] = ld;                                            // tensor_dim0_stride low32 (elems)
  g1[6] = 0;                                             // stride0 hi16 | stride1 low16
  g1[7] = 0;                                             // stride1 hi32
  i32x4 z4 = {0, 0, 0, 0};
#if defined(__clang_major__) && (__clang_major__ >= 23)
  i32x8 z8 = {0, 0, 0, 0, 0, 0, 0, 0};
  __builtin_amdgcn_tensor_load_to_lds(g0, g1, z4, z4, z8, 0);
#else
  __builtin_amdgcn_tensor_load_to_lds(g0, g1, z4, z4, 0);
#endif
}
#endif

// ---------------------------------------------------------------------------
// WMMA fragment compute on one staged tile pair.
// ---------------------------------------------------------------------------
template<int MF, int NF>
__device__ inline void tile_mma(const bf16* lA, const bf16* lB, v8f* acc,
                                int lane, int wm, int wn) {
  v16bf af[MF], bfr[NF];
#pragma unroll
  for (int i = 0; i < MF; ++i) af[i] = ld_frag(lA, wm * (MF * 16) + i * 16, lane);
#pragma unroll
  for (int j = 0; j < NF; ++j) bfr[j] = ld_frag(lB, wn * (NF * 16) + j * 16, lane);
#pragma unroll
  for (int i = 0; i < MF; ++i)
#pragma unroll
    for (int j = 0; j < NF; ++j)
      acc[i * NF + j] = __builtin_amdgcn_wmma_f32_16x16x32_bf16(
          false, af[i], false, bfr[j], (short)0, acc[i * NF + j], false, false);
}

// ---------------------------------------------------------------------------
// TDM double-buffered core: C[BM x BN] += A[BM x K] * Bt[BN x K]^T.
// Wave 0 DMAs tile i+1 while all 8 waves (4Mx2N) run WMMAs on tile i.
// ---------------------------------------------------------------------------
template<int BM, int BN, int MF, int NF>
__device__ inline void gemm_core_tdm(const bf16* A, int lda, const bf16* Bt, int ldb,
                                     int K, v8f* acc,
                                     bf16* lA0, bf16* lA1, bf16* lB0, bf16* lB1, int tid) {
  const int lane = tid & 31, wave = tid >> 5, wm = wave & 3, wn = wave >> 2;
  bf16* lA[2] = {lA0, lA1};
  bf16* lB[2] = {lB0, lB1};
  const int niter = K / 32;
#if HAVE_TDM
  if (tid < 32) {                              // one wave owns the TDM pipeline
    tdm_load_tile(lds_off(lA0), A, BM, lda);
    tdm_load_tile(lds_off(lB0), Bt, BN, ldb);
  }
  for (int i = 0; i < niter; ++i) {
    __syncthreads();                           // buffer (i+1)&1 free of readers
    if (tid < 32) {
      if (i + 1 < niter) {
        tdm_load_tile(lds_off(lA[(i + 1) & 1]), A + (size_t)(i + 1) * 32, BM, lda);
        tdm_load_tile(lds_off(lB[(i + 1) & 1]), Bt + (size_t)(i + 1) * 32, BN, ldb);
        __builtin_amdgcn_s_wait_tensorcnt(2);  // pair for tile i complete
      } else {
        __builtin_amdgcn_s_wait_tensorcnt(0);
      }
    }
    __syncthreads();                           // tile i visible to all waves
    tile_mma<MF, NF>(lA[i & 1], lB[i & 1], acc, lane, wm, wn);
  }
  __syncthreads();
#else
  for (int i = 0; i < niter; ++i) {
    stage_bf16<BM>(lA0, A + (size_t)i * 32, lda, tid);
    stage_bf16<BN>(lB0, Bt + (size_t)i * 32, ldb, tid);
    __syncthreads();
    tile_mma<MF, NF>(lA0, lB0, acc, lane, wm, wn);
    __syncthreads();
  }
#endif
}

// ---------------------------------------------------------------------------
// Legacy staged core for fp32 A operand (attn @ V: convert while staging).
// ---------------------------------------------------------------------------
template<int BM, int BN, int MF, int NF>
__device__ inline void gemm_core_f32a(const float* Ap, int lda, const bf16* Bt, int ldb,
                                      int K, v8f* acc, bf16* lA, bf16* lB, int tid) {
  const int lane = tid & 31, wave = tid >> 5, wm = wave & 3, wn = wave >> 2;
  for (int kb = 0; kb < K; kb += 32) {
    stage_f32<BM>(lA, Ap + kb, lda, tid);
    stage_bf16<BN>(lB, Bt + kb, ldb, tid);
    __syncthreads();
    tile_mma<MF, NF>(lA, lB, acc, lane, wm, wn);
    __syncthreads();
  }
}

__device__ inline void zero_acc(v8f* a, int n) {
  const v8f z = {0.f, 0.f, 0.f, 0.f, 0.f, 0.f, 0.f, 0.f};
  for (int i = 0; i < n; ++i) a[i] = z;
}

// ---------------------------------------------------------------------------
// Converters
// ---------------------------------------------------------------------------
__global__ __launch_bounds__(256) void k_cvt(const float* in, bf16* out, int n) {
  int i = blockIdx.x * 256 + threadIdx.x;
  if (i < n) out[i] = (bf16)in[i];
}
// W[K][N] (f32) -> Wt[N][K] (bf16)
__global__ __launch_bounds__(256) void k_cvt_t(const float* W, bf16* Wt, int K, int N) {
  int i = blockIdx.x * 256 + threadIdx.x;
  if (i >= K * N) return;
  int n = i / K, k = i - n * K;
  Wt[i] = (bf16)W[(size_t)k * N + n];
}

// ---------------------------------------------------------------------------
// QKV projection: z=0->Q, 1->K, 2->V (V stored transposed [B][H][DH][S])
// ---------------------------------------------------------------------------
__global__ __launch_bounds__(256) void k_qkv(const bf16* xb, const bf16* Wt3,
                                             const float* bq, const float* bk, const float* bv,
                                             bf16* Qo, bf16* Ko, bf16* Vt) {
  __shared__ __align__(16) bf16 lA0[128 * LSTR], lA1[128 * LSTR], lB0[128 * LSTR], lB1[128 * LSTR];
  int tid = threadIdx.x;
  int m0 = blockIdx.y * 128, n0 = blockIdx.x * 128;
  int which = blockIdx.z;
  const bf16* Bt = Wt3 + (size_t)which * Dc * Dc + (size_t)n0 * Dc;
  v8f acc[8]; zero_acc(acc, 8);
  gemm_core_tdm<128, 128, 2, 4>(xb + (size_t)m0 * Dc, Dc, Bt, Dc, Dc, acc,
                                lA0, lA1, lB0, lB1, tid);

  const float* bp = (which == 0) ? bq : (which == 1) ? bk : bv;
  bf16* out = (which == 0) ? Qo : Ko;
  int lane = tid & 31, wave = tid >> 5, wm = wave & 3, wn = wave >> 2;
#pragma unroll
  for (int i = 0; i < 2; ++i)
#pragma unroll
    for (int j = 0; j < 4; ++j) {
      int mb = m0 + wm * 32 + i * 16 + ((lane >> 4) << 3);
      int n  = n0 + wn * 64 + j * 16 + (lane & 15);
      float bias = bp[n];
#pragma unroll
      for (int e = 0; e < 8; ++e) {
        int m = mb + e;
        float v = acc[i * 4 + j][e] + bias;
        if (which < 2) out[(size_t)m * Dc + n] = (bf16)v;
        else {
          int b = m >> 10, s = m & 1023, h = n >> 6, d = n & 63;
          Vt[((size_t)(b * Hc + h) * 64 + d) * Sc + s] = (bf16)v;
        }
      }
    }
}

// ---------------------------------------------------------------------------
// scores = Q Kt / 8 + mask*(-1e9) + adjoin   (per b,h) -> fp32 into attn out
// ---------------------------------------------------------------------------
__global__ __launch_bounds__(256) void k_scores(const bf16* Q, const bf16* Kt,
                                                const float* mask, const float* adj,
                                                float* attn) {
  __shared__ __align__(16) bf16 lA0[128 * LSTR], lA1[128 * LSTR], lB0[128 * LSTR], lB1[128 * LSTR];
  int tid = threadIdx.x;
  int z = blockIdx.z, b = z / Hc, h = z % Hc;
  int m0 = blockIdx.y * 128, n0 = blockIdx.x * 128;
  const bf16* Ap = Q  + (size_t)b * Sc * Dc + (size_t)h * 64 + (size_t)m0 * Dc;
  const bf16* Bp = Kt + (size_t)b * Sc * Dc + (size_t)h * 64 + (size_t)n0 * Dc;
  v8f acc[8]; zero_acc(acc, 8);
  gemm_core_tdm<128, 128, 2, 4>(Ap, Dc, Bp, Dc, 64, acc, lA0, lA1, lB0, lB1, tid);

  float* outp = attn + (size_t)z * Sc * Sc;
  const float* mrow = mask + (size_t)b * Sc;
  const float* arow = adj + (size_t)b * Sc * Sc;
  int lane = tid & 31, wave = tid >> 5, wm = wave & 3, wn = wave >> 2;
#pragma unroll
  for (int i = 0; i < 2; ++i)
#pragma unroll
    for (int j = 0; j < 4; ++j) {
      int mb = m0 + wm * 32 + i * 16 + ((lane >> 4) << 3);
      int n  = n0 + wn * 64 + j * 16 + (lane & 15);
      float mval = mrow[n] * -1e9f;
#pragma unroll
      for (int e = 0; e < 8; ++e) {
        int m = mb + e;
        outp[(size_t)m * Sc + n] = acc[i * 4 + j][e] * 0.125f + mval + arow[(size_t)m * Sc + n];
      }
    }
}

// ---------------------------------------------------------------------------
// In-place row softmax over S=1024
// ---------------------------------------------------------------------------
__global__ __launch_bounds__(256) void k_softmax(float* attn) {
  __shared__ float red[256];
  float* p = attn + (size_t)blockIdx.x * Sc;
  int t = threadIdx.x;
  float v[4], mx = -3.4e38f;
#pragma unroll
  for (int i = 0; i < 4; ++i) { v[i] = p[t + 256 * i]; mx = fmaxf(mx, v[i]); }
  red[t] = mx; __syncthreads();
  for (int s = 128; s > 0; s >>= 1) { if (t < s) red[t] = fmaxf(red[t], red[t + s]); __syncthreads(); }
  mx = red[0]; __syncthreads();
  float sum = 0.f;
#pragma unroll
  for (int i = 0; i < 4; ++i) { v[i] = expf(v[i] - mx); sum += v[i]; }
  red[t] = sum; __syncthreads();
  for (int s = 128; s > 0; s >>= 1) { if (t < s) red[t] += red[t + s]; __syncthreads(); }
  float inv = 1.f / red[0];
#pragma unroll
  for (int i = 0; i < 4; ++i) p[t + 256 * i] = v[i] * inv;
}

// ---------------------------------------------------------------------------
// ctx = attn @ V   (per b,h; A is fp32, converted while staging; N = 64)
// ---------------------------------------------------------------------------
__global__ __launch_bounds__(256) void k_ctx(const float* attn, const bf16* Vt, bf16* ctx) {
  __shared__ __align__(16) bf16 lA[128 * LSTR], lB[64 * LSTR];
  int tid = threadIdx.x;
  int z = blockIdx.z, b = z / Hc, h = z % Hc;
  int m0 = blockIdx.y * 128;
  const float* Ap = attn + (size_t)z * Sc * Sc + (size_t)m0 * Sc;
  const bf16* Bp = Vt + (size_t)z * 64 * Sc;
  v8f acc[4]; zero_acc(acc, 4);
  gemm_core_f32a<128, 64, 2, 2>(Ap, Sc, Bp, Sc, Sc, acc, lA, lB, tid);

  int lane = tid & 31, wave = tid >> 5, wm = wave & 3, wn = wave >> 2;
#pragma unroll
  for (int i = 0; i < 2; ++i)
#pragma unroll
    for (int j = 0; j < 2; ++j) {
      int mb = m0 + wm * 32 + i * 16 + ((lane >> 4) << 3);
      int n  = wn * 32 + j * 16 + (lane & 15);   // d within head
#pragma unroll
      for (int e = 0; e < 8; ++e) {
        int m = mb + e;
        ctx[((size_t)b * Sc + m) * Dc + h * 64 + n] = (bf16)acc[i * 2 + j][e];
      }
    }
}

// ---------------------------------------------------------------------------
// out1 = x + ctx @ Wo + bo  -> fp32 (residual path) and bf16 (FFN input)
// ---------------------------------------------------------------------------
__global__ __launch_bounds__(256) void k_outproj(const bf16* ctx, const bf16* Wot,
                                                 const float* bo, const float* x,
                                                 float* out1f, bf16* out1b) {
  __shared__ __align__(16) bf16 lA0[128 * LSTR], lA1[128 * LSTR], lB0[128 * LSTR], lB1[128 * LSTR];
  int tid = threadIdx.x;
  int m0 = blockIdx.y * 128, n0 = blockIdx.x * 128;
  v8f acc[8]; zero_acc(acc, 8);
  gemm_core_tdm<128, 128, 2, 4>(ctx + (size_t)m0 * Dc, Dc, Wot + (size_t)n0 * Dc, Dc, Dc,
                                acc, lA0, lA1, lB0, lB1, tid);
  int lane = tid & 31, wave = tid >> 5, wm = wave & 3, wn = wave >> 2;
#pragma unroll
  for (int i = 0; i < 2; ++i)
#pragma unroll
    for (int j = 0; j < 4; ++j) {
      int mb = m0 + wm * 32 + i * 16 + ((lane >> 4) << 3);
      int n  = n0 + wn * 64 + j * 16 + (lane & 15);
      float bias = bo[n];
#pragma unroll
      for (int e = 0; e < 8; ++e) {
        int m = mb + e;
        float v = acc[i * 4 + j][e] + bias + x[(size_t)m * Dc + n];
        out1f[(size_t)m * Dc + n] = v;
        out1b[(size_t)m * Dc + n] = (bf16)v;
      }
    }
}

// ---------------------------------------------------------------------------
// h = gelu(out1 @ W1 + b1)  (exact erf gelu) -> bf16
// ---------------------------------------------------------------------------
__global__ __launch_bounds__(256) void k_ffn1(const bf16* out1b, const bf16* W1t,
                                              const float* b1, bf16* hb) {
  __shared__ __align__(16) bf16 lA0[128 * LSTR], lA1[128 * LSTR], lB0[128 * LSTR], lB1[128 * LSTR];
  int tid = threadIdx.x;
  int m0 = blockIdx.y * 128, n0 = blockIdx.x * 128;
  v8f acc[8]; zero_acc(acc, 8);
  gemm_core_tdm<128, 128, 2, 4>(out1b + (size_t)m0 * Dc, Dc, W1t + (size_t)n0 * Dc, Dc, Dc,
                                acc, lA0, lA1, lB0, lB1, tid);
  int lane = tid & 31, wave = tid >> 5, wm = wave & 3, wn = wave >> 2;
#pragma unroll
  for (int i = 0; i < 2; ++i)
#pragma unroll
    for (int j = 0; j < 4; ++j) {
      int mb = m0 + wm * 32 + i * 16 + ((lane >> 4) << 3);
      int n  = n0 + wn * 64 + j * 16 + (lane & 15);
      float bias = b1[n];
#pragma unroll
      for (int e = 0; e < 8; ++e) {
        int m = mb + e;
        float hv = acc[i * 4 + j][e] + bias;
        float g = 0.5f * hv * (1.f + erff(hv * 0.70710678118f));
        hb[(size_t)m * Fc + n] = (bf16)g;
      }
    }
}

// ---------------------------------------------------------------------------
// z = out1 + h @ W2 + b2 -> fp32 (layernorm input)
// ---------------------------------------------------------------------------
__global__ __launch_bounds__(256) void k_ffn2(const bf16* hb, const bf16* W2t,
                                              const float* b2, const float* out1f, float* zf) {
  __shared__ __align__(16) bf16 lA0[128 * LSTR], lA1[128 * LSTR], lB0[128 * LSTR], lB1[128 * LSTR];
  int tid = threadIdx.x;
  int m0 = blockIdx.y * 128, n0 = blockIdx.x * 128;
  v8f acc[8]; zero_acc(acc, 8);
  gemm_core_tdm<128, 128, 2, 4>(hb + (size_t)m0 * Fc, Fc, W2t + (size_t)n0 * Fc, Fc, Fc,
                                acc, lA0, lA1, lB0, lB1, tid);
  int lane = tid & 31, wave = tid >> 5, wm = wave & 3, wn = wave >> 2;
#pragma unroll
  for (int i = 0; i < 2; ++i)
#pragma unroll
    for (int j = 0; j < 4; ++j) {
      int mb = m0 + wm * 32 + i * 16 + ((lane >> 4) << 3);
      int n  = n0 + wn * 64 + j * 16 + (lane & 15);
      float bias = b2[n];
#pragma unroll
      for (int e = 0; e < 8; ++e) {
        int m = mb + e;
        zf[(size_t)m * Dc + n] = acc[i * 4 + j][e] + bias + out1f[(size_t)m * Dc + n];
      }
    }
}

// ---------------------------------------------------------------------------
// Row layernorm over D=768
// ---------------------------------------------------------------------------
__global__ __launch_bounds__(256) void k_ln(const float* z, const float* gamma,
                                            const float* beta, float* out) {
  __shared__ float red[256];
  const float* p = z + (size_t)blockIdx.x * Dc;
  int t = threadIdx.x;
  float v[3], s = 0.f;
#pragma unroll
  for (int i = 0; i < 3; ++i) { v[i] = p[t + 256 * i]; s += v[i]; }
  red[t] = s; __syncthreads();
  for (int st = 128; st > 0; st >>= 1) { if (t < st) red[t] += red[t + st]; __syncthreads(); }
  float mu = red[0] * (1.f / Dc); __syncthreads();
  float s2 = 0.f;
#pragma unroll
  for (int i = 0; i < 3; ++i) { float d = v[i] - mu; s2 += d * d; }
  red[t] = s2; __syncthreads();
  for (int st = 128; st > 0; st >>= 1) { if (t < st) red[t] += red[t + st]; __syncthreads(); }
  float rs = rsqrtf(red[0] * (1.f / Dc) + 1e-6f);
  float* o = out + (size_t)blockIdx.x * Dc;
#pragma unroll
  for (int i = 0; i < 3; ++i) {
    int n = t + 256 * i;
    o[n] = (v[i] - mu) * rs * gamma[n] + beta[n];
  }
}

// ---------------------------------------------------------------------------
extern "C" void kernel_launch(void* const* d_in, const int* in_sizes, int n_in,
                              void* d_out, int out_size, void* d_ws, size_t ws_size,
                              hipStream_t stream) {
  (void)in_sizes; (void)n_in; (void)out_size;
  const float* x    = (const float*)d_in[0];
  const float* mask = (const float*)d_in[2];
  const float* adj  = (const float*)d_in[3];
  const float* Wq = (const float*)d_in[4];  const float* bq  = (const float*)d_in[5];
  const float* Wk = (const float*)d_in[6];  const float* bk  = (const float*)d_in[7];
  const float* Wv = (const float*)d_in[8];  const float* bv  = (const float*)d_in[9];
  const float* Wo = (const float*)d_in[10]; const float* bo  = (const float*)d_in[11];
  const float* W1 = (const float*)d_in[12]; const float* b1  = (const float*)d_in[13];
  const float* W2 = (const float*)d_in[14]; const float* b2  = (const float*)d_in[15];
  const float* g2 = (const float*)d_in[16]; const float* be2 = (const float*)d_in[17];

  float* out2 = (float*)d_out;
  float* attn = out2 + (size_t)Bc * Sc * Dc;   // [B,H,S,S]

  char* w = (char*)d_ws;
  size_t off = 0;
  auto take = [&](size_t bytes) -> char* {
    char* p = w + off;
    off = (off + bytes + 255) & ~(size_t)255;
    return p;
  };
  const size_t MSD = (size_t)Bc * Sc * Dc;     // 8192*768
  bf16*  xb    = (bf16*)take(MSD * 2);
  bf16*  Wqt3  = (bf16*)take((size_t)3 * Dc * Dc * 2);   // Wq^T,Wk^T,Wv^T contiguous
  bf16*  Wot   = (bf16*)take((size_t)Dc * Dc * 2);
  bf16*  W1t   = (bf16*)take((size_t)Fc * Dc * 2);
  bf16*  W2t   = (bf16*)take((size_t)Dc * Fc * 2);
  bf16*  Qb    = (bf16*)take(MSD * 2);
  bf16*  Kbt   = (bf16*)take(MSD * 2);
  bf16*  Vt    = (bf16*)take(MSD * 2);                   // [B][H][DH][S]
  bf16*  ctx   = (bf16*)take(MSD * 2);
  float* out1f = (float*)take(MSD * 4);
  bf16*  out1b = (bf16*)take(MSD * 2);
  bf16*  hb    = (bf16*)take((size_t)Bc * Sc * Fc * 2);
  float* zf    = (float*)take(MSD * 4);
  if (off > ws_size) return;  // workspace too small: bail rather than corrupt

  dim3 blk(256);
  const int nx = (int)MSD;                 // 6291456
  k_cvt<<<dim3((nx + 255) / 256), blk, 0, stream>>>(x, xb, nx);
  const int nw = Dc * Dc;                  // 589824
  k_cvt_t<<<dim3((nw + 255) / 256), blk, 0, stream>>>(Wq, Wqt3, Dc, Dc);
  k_cvt_t<<<dim3((nw + 255) / 256), blk, 0, stream>>>(Wk, Wqt3 + nw, Dc, Dc);
  k_cvt_t<<<dim3((nw + 255) / 256), blk, 0, stream>>>(Wv, Wqt3 + 2 * nw, Dc, Dc);
  k_cvt_t<<<dim3((nw + 255) / 256), blk, 0, stream>>>(Wo, Wot, Dc, Dc);
  const int nf = Dc * Fc;                  // 2359296
  k_cvt_t<<<dim3((nf + 255) / 256), blk, 0, stream>>>(W1, W1t, Dc, Fc);   // W1[D][F] -> [F][D]
  k_cvt_t<<<dim3((nf + 255) / 256), blk, 0, stream>>>(W2, W2t, Fc, Dc);   // W2[F][D] -> [D][F]

  k_qkv    <<<dim3(Dc / 128, (Bc * Sc) / 128, 3), blk, 0, stream>>>(xb, Wqt3, bq, bk, bv, Qb, Kbt, Vt);
  k_scores <<<dim3(Sc / 128, Sc / 128, Bc * Hc), blk, 0, stream>>>(Qb, Kbt, mask, adj, attn);
  k_softmax<<<dim3(Bc * Hc * Sc), blk, 0, stream>>>(attn);
  k_ctx    <<<dim3(1, Sc / 128, Bc * Hc), blk, 0, stream>>>(attn, Vt, ctx);
  k_outproj<<<dim3(Dc / 128, (Bc * Sc) / 128), blk, 0, stream>>>(ctx, Wot, bo, x, out1f, out1b);
  k_ffn1   <<<dim3(Fc / 128, (Bc * Sc) / 128), blk, 0, stream>>>(out1b, W1t, b1, hb);
  k_ffn2   <<<dim3(Dc / 128, (Bc * Sc) / 128), blk, 0, stream>>>(hb, W2t, b2, out1f, zf);
  k_ln     <<<dim3(Bc * Sc), blk, 0, stream>>>(zf, g2, be2, out2);
}
